// LeakyIntegrator_68032281969204
// MI455X (gfx1250) — matrix operand order repaired and checked by reading, not compiled, not run on hardware
//
#include <hip/hip_runtime.h>
#include <hip/hip_bf16.h>

// ---------------------------------------------------------------------------
// Leaky-integrator connectome RNN on MI455X (gfx1250).
//   rec_t = Wc @ relu(x_t) + We @ x_t - we_row * x_t
//   x_{t+1} = alpha*(rec_t + bias + u_t - x_t) + x_t
// [Wc | We] folded into one 2048x4096 f16 matrix (scaled 2^16 to keep 1e-6
// magnitudes in f16 normal range), pre-swizzled into v_wmma_f32_16x16x32_f16
// A-fragment order. 128 persistent WGs (one per 16-row tile, 16 waves each);
// each wave's 8 A-fragments are t-invariant -> live in VGPRs for all 2048
// steps (whole 16MB weight matrix pinned in the grid's register file).
// Per step: broadcast-B WMMA GEMV over K=4096, LDS reduce, pointwise update,
// publish v=[relu(x);x] (ping-pong), two-level global generation barrier.
// ---------------------------------------------------------------------------

#define NN      2048
#define TT      2048
#define KTOT    4096          // 2*NN combined K dimension
#define KC      128           // K chunks of 32
#define NTILES  128           // NN / 16 row tiles == grid size
#define NWAVES  16            // 512 threads / 32
#define QPW     8             // K-chunks (WMMAs) per wave per step
#define DT_     0.01f
#define WSCALE  65536.0f
#define INV_WSCALE (1.0f / 65536.0f)

// barrier layout (uint indices into bar[]): gen @0, root count @16,
// group g counter @32+16*g  (64B separation between hot lines)
#define BAR_GEN   0
#define BAR_ROOT  16
#define BAR_GRP   32
#define NGROUPS   8           // 128 WGs -> 8 groups of 16

typedef _Float16 v16h __attribute__((ext_vector_type(16)));
typedef float    v8f  __attribute__((ext_vector_type(8)));
typedef unsigned int u32x4 __attribute__((ext_vector_type(4)));

// --------------------------- setup kernel 1 --------------------------------
// Combined weight matrix in WMMA A-fragment order:
// element index = ((tile*KC + kc)*32 + lane)*16 + e ; A 16x32 k order used here, same as B:
//   M = lane%16,  k = (lane/16)*16 + e
__global__ void li_build_wtiled(const float* __restrict__ sc,
                                const float* __restrict__ se,
                                const float* __restrict__ sg,
                                const float* __restrict__ mc,
                                const float* __restrict__ me,
                                _Float16* __restrict__ Wt) {
    size_t idx = (size_t)blockIdx.x * blockDim.x + threadIdx.x;
    if (idx >= (size_t)NN * KTOT) return;
    int e  = (int)(idx & 15);
    int l  = (int)((idx >> 4) & 31);
    int kc = (int)((idx >> 9) & (KC - 1));
    int g  = (int)(idx >> 16);                 // 16*32*128 = 65536 elems/tile
    int i  = g * 16 + (l & 15);
    int kb = (l >> 4) * 16;
    int k  = kb + e;
    int col = kc * 32 + k;
    float w;
    if (col < NN) {                            // chemical half
        size_t o = (size_t)i * NN + col;
        float m = mc[o]; m = m > 0.f ? m : 0.f;
        w = sc[o] * m * sg[o];
    } else {                                   // electrical half (symmetric)
        int j = col - NN;
        float a = me[(size_t)i * NN + j]; a = a > 0.f ? a : 0.f;
        float b = me[(size_t)j * NN + i]; b = b > 0.f ? b : 0.f;
        w = se[(size_t)i * NN + j] * (a + b);
    }
    Wt[idx] = (_Float16)(w * WSCALE);
}

// --------------------------- setup kernel 2 --------------------------------
__global__ void li_setup_rows(const float* __restrict__ se,
                              const float* __restrict__ me,
                              const float* __restrict__ tau,
                              const float* __restrict__ h0,
                              float* __restrict__ we_row,
                              float* __restrict__ alphav,
                              _Float16* __restrict__ vbuf) {
    int i = blockIdx.x;
    __shared__ float red[256];
    float s = 0.f;
    for (int j = threadIdx.x; j < NN; j += 256) {
        float a = me[(size_t)i * NN + j]; a = a > 0.f ? a : 0.f;
        float b = me[(size_t)j * NN + i]; b = b > 0.f ? b : 0.f;
        s += se[(size_t)i * NN + j] * (a + b);
    }
    red[threadIdx.x] = s;
    __syncthreads();
    for (int off = 128; off > 0; off >>= 1) {
        if (threadIdx.x < off) red[threadIdx.x] += red[threadIdx.x + off];
        __syncthreads();
    }
    if (threadIdx.x == 0) {
        we_row[i] = red[0];
        float tc = tau[i]; tc = tc > DT_ ? tc : DT_;
        alphav[i] = DT_ / tc;
        float x = h0[i];
        vbuf[i]      = (_Float16)(x > 0.f ? x : 0.f);   // relu(x) half
        vbuf[NN + i] = (_Float16)x;                     // linear half
    }
}

// --------------------------- main persistent kernel ------------------------
__global__ __launch_bounds__(512, 1)
void li_persistent(const float* __restrict__ input,
                   const float* __restrict__ bias,
                   const _Float16* __restrict__ Wt,
                   const float* __restrict__ we_row,
                   const float* __restrict__ alphav,
                   _Float16* __restrict__ vbuf,      // 2 ping-pong buffers of KTOT f16
                   unsigned* __restrict__ bar,
                   float* __restrict__ out_hidden,
                   float* __restrict__ out_rec,
                   float* __restrict__ out_ext,
                   const float* __restrict__ h0) {
    __shared__ __align__(32) _Float16 vlds[KTOT];     // 8 KB: [relu(x); x] in f16
    __shared__ float partial[NWAVES][16];
    __shared__ float xloc[16];

    const int tid  = threadIdx.x;
    const int g    = blockIdx.x;          // row-tile id, rows [g*16, g*16+16)
    const int wave = tid >> 5;
    const int lane = tid & 31;
    const int i0   = g * 16;
    const int gi   = g >> 4;              // barrier group (8 groups of 16 WGs)

    // t-invariant per-neuron constants (threads 0..15 only)
    float my_we = 0.f, my_bias = 0.f, my_alpha = 0.f;
    if (tid < 16) {
        xloc[tid] = h0[i0 + tid];
        my_we    = we_row[i0 + tid];
        my_bias  = bias[i0 + tid];
        my_alpha = alphav[i0 + tid];
    }

    // Preload this wave's 8 A-fragments: t-invariant, pinned in VGPRs.
    const _Float16* Abase = Wt + (size_t)g * (KC * 32 * 16);
    const int kc0 = wave * QPW;
    v16h af[QPW];
#pragma unroll
    for (int q = 0; q < QPW; ++q)
        af[q] = *(const v16h*)(Abase + (((size_t)(kc0 + q) * 32 + lane) << 4));

    const int kb = (lane >> 4) * 16;      // B-fragment K offset for this half-wave
    const _Float16* bsrc = vlds + kc0 * 32 + kb;

    for (int t = 0; t < TT; ++t) {
        const _Float16* vcur = vbuf + (size_t)(t & 1) * KTOT;
        _Float16*       vnxt = vbuf + (size_t)((t + 1) & 1) * KTOT;

        // stage v_t into LDS (512 threads x 16B); issue input load early so
        // its latency hides under the WMMA chain.
        ((u32x4*)vlds)[tid] = ((const u32x4*)vcur)[tid];
        float u = 0.f;
        if (tid < 16) u = input[(size_t)t * NN + i0 + tid];
        __syncthreads();

        // GEMV: batch B-fragment LDS loads 4-deep, chain 8 WMMAs (f32 acc).
        v8f acc = {0.f, 0.f, 0.f, 0.f, 0.f, 0.f, 0.f, 0.f};
        {
            v16h b0 = *(const v16h*)(bsrc + 0 * 32);
            v16h b1 = *(const v16h*)(bsrc + 1 * 32);
            v16h b2 = *(const v16h*)(bsrc + 2 * 32);
            v16h b3 = *(const v16h*)(bsrc + 3 * 32);
            acc = __builtin_amdgcn_wmma_f32_16x16x32_f16(false, af[0], false, b0, (short)0, acc, false, false);
            acc = __builtin_amdgcn_wmma_f32_16x16x32_f16(false, af[1], false, b1, (short)0, acc, false, false);
            v16h b4 = *(const v16h*)(bsrc + 4 * 32);
            v16h b5 = *(const v16h*)(bsrc + 5 * 32);
            acc = __builtin_amdgcn_wmma_f32_16x16x32_f16(false, af[2], false, b2, (short)0, acc, false, false);
            acc = __builtin_amdgcn_wmma_f32_16x16x32_f16(false, af[3], false, b3, (short)0, acc, false, false);
            v16h b6 = *(const v16h*)(bsrc + 6 * 32);
            v16h b7 = *(const v16h*)(bsrc + 7 * 32);
            acc = __builtin_amdgcn_wmma_f32_16x16x32_f16(false, af[4], false, b4, (short)0, acc, false, false);
            acc = __builtin_amdgcn_wmma_f32_16x16x32_f16(false, af[5], false, b5, (short)0, acc, false, false);
            acc = __builtin_amdgcn_wmma_f32_16x16x32_f16(false, af[6], false, b6, (short)0, acc, false, false);
            acc = __builtin_amdgcn_wmma_f32_16x16x32_f16(false, af[7], false, b7, (short)0, acc, false, false);
        }
        // C layout: lane l, vgpr i -> row (i + 8*(l/16)); all 16 cols identical.
        if ((lane & 15) == 0) {
            const int rbase = (lane >> 4) * 8;
#pragma unroll
            for (int i = 0; i < 8; ++i) partial[wave][rbase + i] = acc[i];
        }
        __syncthreads();

        // pointwise state update for this tile's 16 neurons
        if (tid < 16) {
            float y = 0.f;
#pragma unroll
            for (int w = 0; w < NWAVES; ++w) y += partial[w][tid];
            y *= INV_WSCALE;
            const size_t o = (size_t)t * NN + i0 + tid;
            const float x_old = xloc[tid];
            const float rec = y - my_we * x_old;
            const float xn = my_alpha * (rec + my_bias + u - x_old) + x_old;
            out_hidden[o] = xn;
            out_rec[o]    = rec;
            out_ext[o]    = u;
            xloc[tid] = xn;
            vnxt[i0 + tid]      = (_Float16)(xn > 0.f ? xn : 0.f);
            vnxt[NN + i0 + tid] = (_Float16)xn;
        }

        // ---- two-level grid barrier (monotonic generations, no resets) ----
        __syncthreads();
        if (tid == 0) {
            __threadfence();
            const unsigned gen_target = (unsigned)(t + 1);
            const unsigned a = atomicAdd(&bar[BAR_GRP + 16 * gi], 1u) + 1u;
            if (a == 16u * gen_target) {                       // last in group
                const unsigned r = atomicAdd(&bar[BAR_ROOT], 1u) + 1u;
                if (r == (unsigned)NGROUPS * gen_target)       // last overall
                    __hip_atomic_store(&bar[BAR_GEN], gen_target,
                                       __ATOMIC_RELEASE, __HIP_MEMORY_SCOPE_AGENT);
            }
            while (__hip_atomic_load(&bar[BAR_GEN], __ATOMIC_ACQUIRE,
                                     __HIP_MEMORY_SCOPE_AGENT) < gen_target)
                __builtin_amdgcn_s_sleep(1);
            __threadfence();
        }
        __syncthreads();
    }
}

// --------------------------- host launcher ---------------------------------
extern "C" void kernel_launch(void* const* d_in, const int* in_sizes, int n_in,
                              void* d_out, int out_size, void* d_ws, size_t ws_size,
                              hipStream_t stream) {
    (void)in_sizes; (void)n_in; (void)out_size; (void)ws_size;
    const float* input = (const float*)d_in[0];
    const float* sc    = (const float*)d_in[1];
    const float* se    = (const float*)d_in[2];
    const float* sg    = (const float*)d_in[3];
    const float* mc    = (const float*)d_in[4];
    const float* me    = (const float*)d_in[5];
    const float* bias  = (const float*)d_in[6];
    const float* tau   = (const float*)d_in[7];
    const float* h0    = (const float*)d_in[8];

    char* ws = (char*)d_ws;
    _Float16* Wt = (_Float16*)ws;                                  // 16 MiB
    size_t off = (size_t)NN * KTOT * sizeof(_Float16);
    float* we_row = (float*)(ws + off); off += NN * sizeof(float);
    float* alphav = (float*)(ws + off); off += NN * sizeof(float);
    _Float16* vbuf = (_Float16*)(ws + off); off += 2 * KTOT * sizeof(_Float16);
    off = (off + 255) & ~(size_t)255;
    unsigned* bar = (unsigned*)(ws + off);

    float* out_hidden = (float*)d_out;
    float* out_rec    = out_hidden + (size_t)TT * NN;
    float* out_ext    = out_rec    + (size_t)TT * NN;

    hipMemsetAsync(bar, 0, 1024, stream);

    const long total = (long)NN * KTOT;
    li_build_wtiled<<<(unsigned)((total + 255) / 256), 256, 0, stream>>>(
        sc, se, sg, mc, me, Wt);
    li_setup_rows<<<NN, 256, 0, stream>>>(se, me, tau, h0, we_row, alphav, vbuf);
    li_persistent<<<NTILES, 512, 0, stream>>>(
        input, bias, Wt, we_row, alphav, vbuf, bar,
        out_hidden, out_rec, out_ext, h0);
}
